// ChunkedCausalAttention_StaticRoutedLoRAExpert_46737834115825
// MI455X (gfx1250) — compile-verified, matrix operand order checked by reading, not run on hardware
//
#include <hip/hip_runtime.h>
#include <stdint.h>

// ---------------------------------------------------------------------------
// ChunkedCausalAttention + static-routed LoRA experts, CDNA5 (gfx1250) WMMA.
//
// Shapes (from reference): B=4, N=2048 (4 chunks x 512), DIM=1024, H=16, D=64,
// rank=16, LORA_SCALE=2.0, qkv mask = (1,0,1) over the three 1024-col groups.
//
// All heavy math runs through v_wmma_f32_16x16x32_bf16 (fp32 accumulate).
// fp32 inputs are converted once to bf16 in workspace.
//
// WMMA 16x16x32 bf16 operand layout assumed (ISA 7.12.2):
//   A (16x32, MxK): lane l -> row M=l%16; half h -> K = h%8 + (h/8)*16 + (l/16)*8
//   B (32x16, KxN): lane l -> col N=l%16; half h -> K = h%8 + (h/8)*16 + (l/16)*8
//   C/D (16x16):    lane l, vgpr v -> M = v + 8*(l/16), N = l%16
// ---------------------------------------------------------------------------

typedef __bf16 bf16;
typedef __attribute__((ext_vector_type(16))) __bf16 v16bf;
typedef __attribute__((ext_vector_type(8)))  float  v8f;

union FragOp  { v16bf v; uint4 q[2]; };
union FragAcc { v8f   v; float f[8]; };

#define LORA_SCALE_F 2.0f

__device__ __forceinline__ v8f wmma_bf16(v16bf a, v16bf b, v8f c) {
    return __builtin_amdgcn_wmma_f32_16x16x32_bf16(
        /*neg_a=*/false, a, /*neg_b=*/false, b,
        /*c_mod=*/(short)0, c, /*reuse_a=*/false, /*reuse_b=*/false);
}

// ---------------------------------------------------------------------------
// fp32 -> bf16 conversion (x4 vectorized; every tensor size is a multiple of 4)
// ---------------------------------------------------------------------------
__global__ __launch_bounds__(256)
void cvt_f32_bf16_v4(const float* __restrict__ src, bf16* __restrict__ dst,
                     unsigned long long n4) {
    unsigned long long i = (unsigned long long)blockIdx.x * blockDim.x + threadIdx.x;
    if (i < n4) {
        float4 f = reinterpret_cast<const float4*>(src)[i];
        union { bf16 h[4]; uint2 u; } o;
        o.h[0] = (bf16)f.x; o.h[1] = (bf16)f.y;
        o.h[2] = (bf16)f.z; o.h[3] = (bf16)f.w;
        reinterpret_cast<uint2*>(dst)[i] = o.u;
    }
}

// ---------------------------------------------------------------------------
// lora_t[m, r] = sum_k A[m,k] * LA[e(m)][r][k]      (M x K) @ (16 x K)^T
// One wave computes a 16x16 tile over the full K reduction. 4 waves / block.
// ---------------------------------------------------------------------------
__global__ __launch_bounds__(128)
void lora_t_wmma(const bf16* __restrict__ A, const bf16* __restrict__ LA,
                 const int* __restrict__ eidx, float* __restrict__ T, int K) {
    const int wave = threadIdx.x >> 5, lane = threadIdx.x & 31;
    const int l2 = lane & 15, hi = lane >> 4;
    const int row0 = blockIdx.x * 64 + wave * 16;
    const int e = eidx[(row0 & 2047) >> 9];

    FragAcc acc;
#pragma unroll
    for (int t = 0; t < 8; ++t) acc.f[t] = 0.f;

    const bf16* Arow = A  + (size_t)(row0 + l2) * K;
    const bf16* Lrow = LA + ((size_t)e * 16 + l2) * K;
    for (int k0 = 0; k0 < K; k0 += 32) {
        FragOp a, b;
        a.q[0] = *reinterpret_cast<const uint4*>(Arow + k0 + hi * 8);
        a.q[1] = *reinterpret_cast<const uint4*>(Arow + k0 + 16 + hi * 8);
        b.q[0] = *reinterpret_cast<const uint4*>(Lrow + k0 + hi * 8);
        b.q[1] = *reinterpret_cast<const uint4*>(Lrow + k0 + 16 + hi * 8);
        acc.v = wmma_bf16(a.v, b.v, acc.v);
    }
#pragma unroll
    for (int v = 0; v < 8; ++v)
        T[(size_t)(row0 + v + 8 * hi) * 16 + l2] = acc.f[v];
}

// ---------------------------------------------------------------------------
// C[m,n] = sum_k A[m,k]*W[n,k] + mult(n) * sum_r LT[m,r]*LB[e][n,r]
// A: MxK bf16 row-major, W: NxK bf16 row-major, LT: Mx16 f32, LB: [4][N][16] bf16.
// Block tile 128x64, 256 threads = 8 waves (4 along M, 2 along N), 32x32/wave.
// maskmode: 1 = LoRA everywhere; 2 = LoRA masked by (1,0,1) per 1024-col group.
// ---------------------------------------------------------------------------
__global__ __launch_bounds__(256)
void gemm_lora_wmma(const bf16* __restrict__ A, const bf16* __restrict__ W,
                    int M, int N, int K,
                    const float* __restrict__ LT, const bf16* __restrict__ LB,
                    const int* __restrict__ eidx, int maskmode,
                    float* __restrict__ outF, bf16* __restrict__ outB) {
    __shared__ __align__(16) bf16 sA[128 * 40];
    __shared__ __align__(16) bf16 sW[64 * 40];

    const int tid = threadIdx.x;
    const int wave = tid >> 5, lane = tid & 31;
    const int l2 = lane & 15, hi = lane >> 4;
    const int row0 = blockIdx.y * 128;
    const int col0 = blockIdx.x * 64;
    const int wm = wave & 3, wn = wave >> 2;
    const int e = eidx[(row0 & 2047) >> 9];

    FragAcc acc[2][2];
#pragma unroll
    for (int i = 0; i < 2; ++i)
#pragma unroll
        for (int j = 0; j < 2; ++j)
#pragma unroll
            for (int t = 0; t < 8; ++t) acc[i][j].f[t] = 0.f;

    for (int k0 = 0; k0 < K; k0 += 32) {
        // Stage A tile (128x32 halves) and W tile (64x32 halves) into LDS.
#pragma unroll
        for (int j = 0; j < 2; ++j) {
            int idx = tid + j * 256;
            int r = idx >> 2, seg = idx & 3;
            const bf16* gp = A + (size_t)(row0 + r) * K + k0 + seg * 8;
            uint4 d = *reinterpret_cast<const uint4*>(gp);
            if (k0 + 32 < K) __builtin_prefetch(gp + 32, 0, 1);  // global_prefetch_b8
            *reinterpret_cast<uint4*>(&sA[r * 40 + seg * 8]) = d;
        }
        {
            int r = tid >> 2, seg = tid & 3;
            uint4 d = *reinterpret_cast<const uint4*>(
                W + (size_t)(col0 + r) * K + k0 + seg * 8);
            *reinterpret_cast<uint4*>(&sW[r * 40 + seg * 8]) = d;
        }
        __syncthreads();

        FragOp a[2], b[2];
#pragma unroll
        for (int i = 0; i < 2; ++i) {
            const bf16* p = &sA[(wm * 32 + i * 16 + l2) * 40 + hi * 8];
            a[i].q[0] = *reinterpret_cast<const uint4*>(p);
            a[i].q[1] = *reinterpret_cast<const uint4*>(p + 16);
        }
#pragma unroll
        for (int j = 0; j < 2; ++j) {
            const bf16* p = &sW[(wn * 32 + j * 16 + l2) * 40 + hi * 8];
            b[j].q[0] = *reinterpret_cast<const uint4*>(p);
            b[j].q[1] = *reinterpret_cast<const uint4*>(p + 16);
        }
#pragma unroll
        for (int i = 0; i < 2; ++i)
#pragma unroll
            for (int j = 0; j < 2; ++j)
                acc[i][j].v = wmma_bf16(a[i].v, b[j].v, acc[i][j].v);
        __syncthreads();
    }

    // Epilogue: rank-16 LoRA correction + store.
#pragma unroll
    for (int j = 0; j < 2; ++j) {
        const int cj = col0 + wn * 32 + j * 16 + l2;
        float mult = LORA_SCALE_F;
        if (maskmode == 2 && ((cj >> 10) == 1)) mult = 0.f;

        FragOp blr;
        const bf16* pb = LB + ((size_t)e * N + cj) * 16;
        blr.q[0] = *reinterpret_cast<const uint4*>(pb);
        blr.q[1] = *reinterpret_cast<const uint4*>(pb + 8);
        float bl[16];
#pragma unroll
        for (int t = 0; t < 16; ++t) bl[t] = (float)blr.v[t];

#pragma unroll
        for (int i = 0; i < 2; ++i) {
#pragma unroll
            for (int v = 0; v < 8; ++v) {
                const int m = row0 + wm * 32 + i * 16 + v + 8 * hi;
                const float* lt = LT + (size_t)m * 16;
                float s = 0.f;
#pragma unroll
                for (int t = 0; t < 16; ++t) s += lt[t] * bl[t];
                const float val = acc[i][j].f[v] + mult * s;
                const size_t off = (size_t)m * N + cj;
                if (outF) outF[off] = val;
                if (outB) outB[off] = (bf16)val;
            }
        }
    }
}

// ---------------------------------------------------------------------------
// Chunk-causal flash attention over QKV[b, n, 3*1024] (bf16).
// Chunk i's queries attend to the full prefix L=(i+1)*512 keys (no intra mask).
// Block: 256 thr = 8 waves, each wave owns 16 q rows (q-tile = 128 rows).
// grid = (chunk*4 + qtile, head, batch).
// K frags load straight from global; V staged transposed in LDS; P re-layout
// through per-wave LDS to match the WMMA A-operand pattern.
// ---------------------------------------------------------------------------
__global__ __launch_bounds__(256)
void attn_flash_wmma(const bf16* __restrict__ QKV, bf16* __restrict__ O) {
    __shared__ __align__(16) bf16 vt[64 * 40];       // V^T tile: [d][key]
    __shared__ __align__(16) bf16 pl[8 * 16 * 40];   // per-wave P tile 16x32

    const int tid = threadIdx.x, wave = tid >> 5, lane = tid & 31;
    const int l2 = lane & 15, hi = lane >> 4;
    const int b = blockIdx.z, h = blockIdx.y;
    const int chunk = blockIdx.x >> 2, qt = blockIdx.x & 3;
    const int L = (chunk + 1) * 512;
    const int qrow0 = chunk * 512 + qt * 128 + wave * 16;
    const int ld = 3072;
    const float sc = 0.125f;                          // 1/sqrt(64)
    const float L2E = 1.44269504088896340736f;

    // Q fragments (held in registers for the whole KV scan).
    FragOp qf[2];
#pragma unroll
    for (int dc = 0; dc < 2; ++dc) {
        const bf16* p = QKV + ((size_t)b * 2048 + qrow0 + l2) * ld
                        + h * 64 + dc * 32 + hi * 8;
        qf[dc].q[0] = *reinterpret_cast<const uint4*>(p);
        qf[dc].q[1] = *reinterpret_cast<const uint4*>(p + 16);
    }

    float mrow[8], lrow[8];
    FragAcc o[4];
#pragma unroll
    for (int v = 0; v < 8; ++v) { mrow[v] = -1e30f; lrow[v] = 0.f; }
#pragma unroll
    for (int nf = 0; nf < 4; ++nf)
#pragma unroll
        for (int t = 0; t < 8; ++t) o[nf].f[t] = 0.f;

    bf16* myp = &pl[wave * 16 * 40];

    for (int k0 = 0; k0 < L; k0 += 32) {
        // Stage V^T (32 keys x 64 d) into LDS, transposed.
        {
            const int key = tid >> 3, dseg = tid & 7;
            const bf16* pv = QKV + ((size_t)b * 2048 + k0 + key) * ld
                             + 2048 + h * 64 + dseg * 8;
            uint4 d = *reinterpret_cast<const uint4*>(pv);
            const bf16* hv = reinterpret_cast<const bf16*>(&d);
#pragma unroll
            for (int j = 0; j < 8; ++j) vt[(dseg * 8 + j) * 40 + key] = hv[j];
        }
        __syncthreads();

        // S = Q K^T  (16 q x 32 keys), K frags direct from global.
        FragAcc s[2];
#pragma unroll
        for (int nf = 0; nf < 2; ++nf) {
#pragma unroll
            for (int t = 0; t < 8; ++t) s[nf].f[t] = 0.f;
#pragma unroll
            for (int dc = 0; dc < 2; ++dc) {
                FragOp kf;
                const bf16* p = QKV + ((size_t)b * 2048 + k0 + nf * 16 + l2) * ld
                                + 1024 + h * 64 + dc * 32 + hi * 8;
                kf.q[0] = *reinterpret_cast<const uint4*>(p);
                kf.q[1] = *reinterpret_cast<const uint4*>(p + 16);
                s[nf].v = wmma_bf16(qf[dc].v, kf.v, s[nf].v);
            }
        }

        // Online softmax per accumulator row (row = v + 8*hi, col = l2).
#pragma unroll
        for (int v = 0; v < 8; ++v) {
            const float s0 = s[0].f[v] * sc, s1 = s[1].f[v] * sc;
            float mx = fmaxf(s0, s1);
#pragma unroll
            for (int mk = 8; mk >= 1; mk >>= 1)
                mx = fmaxf(mx, __shfl_xor(mx, mk, 32));
            const float mnew  = fmaxf(mrow[v], mx);
            const float alpha = exp2f((mrow[v] - mnew) * L2E);
            const float p0    = exp2f((s0 - mnew) * L2E);
            const float p1    = exp2f((s1 - mnew) * L2E);
            float rs = p0 + p1;
#pragma unroll
            for (int mk = 8; mk >= 1; mk >>= 1)
                rs += __shfl_xor(rs, mk, 32);
            lrow[v] = lrow[v] * alpha + rs;
            mrow[v] = mnew;
#pragma unroll
            for (int nf = 0; nf < 4; ++nf) o[nf].f[v] *= alpha;
            const int prow = v + 8 * hi;
            myp[prow * 40 + l2]      = (bf16)p0;
            myp[prow * 40 + 16 + l2] = (bf16)p1;
        }

        // Re-load P in A-operand layout (same-wave LDS ops stay in order).
        FragOp pf;
        {
            const bf16* pp = &myp[l2 * 40 + hi * 8];
            pf.q[0] = *reinterpret_cast<const uint4*>(pp);
            pf.q[1] = *reinterpret_cast<const uint4*>(pp + 16);
        }

        // O += P V   (contraction over 32 keys, 4 N-frags cover d=64).
#pragma unroll
        for (int nf = 0; nf < 4; ++nf) {
            FragOp vf;
            const bf16* pv = &vt[(nf * 16 + l2) * 40 + hi * 8];
            vf.q[0] = *reinterpret_cast<const uint4*>(pv);
            vf.q[1] = *reinterpret_cast<const uint4*>(pv + 16);
            o[nf].v = wmma_bf16(pf.v, vf.v, o[nf].v);
        }
        __syncthreads();
    }

    // Normalize and write O (bf16) at [b, n, h*64 + d].
#pragma unroll
    for (int v = 0; v < 8; ++v) {
        const float inv = 1.f / lrow[v];
        const size_t r = (size_t)b * 2048 + qrow0 + v + 8 * hi;
#pragma unroll
        for (int nf = 0; nf < 4; ++nf)
            O[r * 1024 + h * 64 + nf * 16 + l2] = (bf16)(o[nf].f[v] * inv);
    }
}

// ---------------------------------------------------------------------------
// Host-side orchestration
// ---------------------------------------------------------------------------
extern "C" void kernel_launch(void* const* d_in, const int* in_sizes, int n_in,
                              void* d_out, int out_size, void* d_ws, size_t ws_size,
                              hipStream_t stream) {
    (void)in_sizes; (void)n_in; (void)out_size; (void)ws_size;

    const float* x  = (const float*)d_in[0];   // [4,2048,1024]
    const float* Wq = (const float*)d_in[1];   // [3072,1024]
    const float* Aq = (const float*)d_in[2];   // [4,16,1024]
    const float* Bq = (const float*)d_in[3];   // [4,3072,16]
    const float* Wp = (const float*)d_in[4];   // [1024,1024]
    const float* Ap = (const float*)d_in[5];   // [4,16,1024]
    const float* Bp = (const float*)d_in[6];   // [4,1024,16]
    const int*   ei = (const int*)d_in[7];     // [4]
    float* out = (float*)d_out;                // [4,2048,1024] fp32

    const int M = 8192;          // B*N rows
    const int K = 1024;          // DIM
    const int Nqkv = 3072;

    // Workspace carve-out (256B aligned blocks).
    char* ws = (char*)d_ws;
    auto alloc = [&](size_t bytes) -> char* {
        char* p = ws;
        ws += (bytes + 255) & ~(size_t)255;
        return p;
    };
    bf16* Xb   = (bf16*)alloc((size_t)M * K * 2);            // 16 MB
    bf16* Wqb  = (bf16*)alloc((size_t)Nqkv * K * 2);         // 6 MB
    bf16* Wpb  = (bf16*)alloc((size_t)K * K * 2);            // 2 MB
    bf16* Aqb  = (bf16*)alloc((size_t)4 * 16 * K * 2);
    bf16* Bqb  = (bf16*)alloc((size_t)4 * Nqkv * 16 * 2);
    bf16* Apb  = (bf16*)alloc((size_t)4 * 16 * K * 2);
    bf16* Bpb  = (bf16*)alloc((size_t)4 * K * 16 * 2);
    float* T1  = (float*)alloc((size_t)M * 16 * 4);
    float* T2  = (float*)alloc((size_t)M * 16 * 4);
    bf16* QKVb = (bf16*)alloc((size_t)M * Nqkv * 2);         // 48 MB
    bf16* Ob   = (bf16*)alloc((size_t)M * K * 2);            // 16 MB

    auto cvt = [&](const float* s, bf16* d, size_t n) {
        size_t n4 = n / 4;
        cvt_f32_bf16_v4<<<dim3((unsigned)((n4 + 255) / 256)), dim3(256), 0, stream>>>(
            s, d, (unsigned long long)n4);
    };
    cvt(x,  Xb,  (size_t)M * K);
    cvt(Wq, Wqb, (size_t)Nqkv * K);
    cvt(Wp, Wpb, (size_t)K * K);
    cvt(Aq, Aqb, (size_t)4 * 16 * K);
    cvt(Bq, Bqb, (size_t)4 * Nqkv * 16);
    cvt(Ap, Apb, (size_t)4 * 16 * K);
    cvt(Bp, Bpb, (size_t)4 * K * 16);

    // 1) T1 = X @ A_e^T
    lora_t_wmma<<<dim3(M / 64), dim3(128), 0, stream>>>(Xb, Aqb, ei, T1, K);

    // 2) QKV = X @ Wq^T + 2.0 * mask * (T1 @ Bq_e^T)   -> bf16
    gemm_lora_wmma<<<dim3(Nqkv / 64, M / 128), dim3(256), 0, stream>>>(
        Xb, Wqb, M, Nqkv, K, T1, Bqb, ei, /*maskmode=*/2, nullptr, QKVb);

    // 3) Chunk-causal flash attention -> Ob (bf16)
    attn_flash_wmma<<<dim3(16, 16, 4), dim3(256), 0, stream>>>(QKVb, Ob);

    // 4) T2 = O @ Ap_e^T
    lora_t_wmma<<<dim3(M / 64), dim3(128), 0, stream>>>(Ob, Apb, ei, T2, K);

    // 5) out = O @ Wp^T + 2.0 * (T2 @ Bp_e^T)          -> fp32
    gemm_lora_wmma<<<dim3(K / 64, M / 128), dim3(256), 0, stream>>>(
        Ob, Wpb, M, K, K, T2, Bpb, ei, /*maskmode=*/1, out, nullptr);
}
